// GAT_Layer_30751965839669
// MI455X (gfx1250) — compile-verified, matrix operand order checked by reading, not run on hardware
//
#include <hip/hip_runtime.h>

#define LEAKY 0.2f
#define EPS_DEN 1e-16f

typedef __attribute__((ext_vector_type(16))) _Float16 v16h;
typedef __attribute__((ext_vector_type(8)))  _Float16 v8h;
typedef __attribute__((ext_vector_type(8)))  float    v8f;

#define WT_STRIDE 136  // halfs; 272 B row pitch -> bank-decorrelated column reads

// ---------------------------------------------------------------- utilities

__global__ __launch_bounds__(256)
void fill_f32(float* __restrict__ p, float v, long long n) {
  long long i = (long long)blockIdx.x * blockDim.x + threadIdx.x;
  if (i < n) p[i] = v;
}

__device__ __forceinline__ void atomicMaxFloat(float* addr, float val) {
  // Sign-split trick: exact float max via integer atomics (init must be -inf).
  if (val >= 0.0f) {
    atomicMax((int*)addr, __float_as_int(val));
  } else {
    atomicMin((unsigned int*)addr, __float_as_uint(val));
  }
}

// ------------------------------------ fused dual WMMA GEMM  Y1=X@W1, Y2=X@W2
// X: [nrows,128] f32 row-major; W1,W2: [128,128] f32 row-major.
// 256 threads = 8 waves/block; each wave owns one 16-row M-tile, loads its A
// fragments ONCE (4 K-chunks of v_wmma_f32_16x16x32_f16), then sweeps 8 N-tiles
// for each of the two weight matrices. Weights are staged transposed in LDS as
// f16 so every B fragment is two contiguous ds_load_b128 per lane.
__global__ __launch_bounds__(256)
void wmma_dual_gemm_128(const float* __restrict__ X,
                        const float* __restrict__ W1,
                        const float* __restrict__ W2,
                        float* __restrict__ Y1, float* __restrict__ Y2,
                        int nrows) {
  __shared__ _Float16 Wt[2][128 * WT_STRIDE];  // 68 KB: Wt[w][n*136 + k] = W[k][n]
  const int t = threadIdx.x;
  #pragma unroll
  for (int w = 0; w < 2; ++w) {
    const float* Wsrc = w ? W2 : W1;
    for (int i = 0; i < 64; ++i) {            // coalesced read, transposed store
      const int idx = t + 256 * i;
      const int k = idx >> 7;
      const int n = idx & 127;
      Wt[w][n * WT_STRIDE + k] = (_Float16)Wsrc[idx];
    }
  }
  __syncthreads();

  const int wave  = t >> 5;
  const int lane  = t & 31;
  const int m16   = lane & 15;   // row (A) / col (B,D) within tile
  const int hi    = lane >> 4;   // upper-half lane group
  const int tileM = blockIdx.x * 8 + wave;
  if (tileM * 16 >= nrows) return;           // wave-uniform guard

  const float* xrow = X + (size_t)(tileM * 16 + m16) * 128;

  // A fragments: ISA 16-bit 16x32 layout -> elements 0..7 are K=k0+8hi+{0..7},
  // elements 8..15 are K=k0+8hi+16+{0..7}: four float4 loads + pack-convert.
  v16h afrag[4];
  #pragma unroll
  for (int kc = 0; kc < 4; ++kc) {
    const int k0 = kc * 32 + 8 * hi;
    const float4 f0 = *reinterpret_cast<const float4*>(xrow + k0);
    const float4 f1 = *reinterpret_cast<const float4*>(xrow + k0 + 4);
    const float4 f2 = *reinterpret_cast<const float4*>(xrow + k0 + 16);
    const float4 f3 = *reinterpret_cast<const float4*>(xrow + k0 + 20);
    afrag[kc][0]  = (_Float16)f0.x; afrag[kc][1]  = (_Float16)f0.y;
    afrag[kc][2]  = (_Float16)f0.z; afrag[kc][3]  = (_Float16)f0.w;
    afrag[kc][4]  = (_Float16)f1.x; afrag[kc][5]  = (_Float16)f1.y;
    afrag[kc][6]  = (_Float16)f1.z; afrag[kc][7]  = (_Float16)f1.w;
    afrag[kc][8]  = (_Float16)f2.x; afrag[kc][9]  = (_Float16)f2.y;
    afrag[kc][10] = (_Float16)f2.z; afrag[kc][11] = (_Float16)f2.w;
    afrag[kc][12] = (_Float16)f3.x; afrag[kc][13] = (_Float16)f3.y;
    afrag[kc][14] = (_Float16)f3.z; afrag[kc][15] = (_Float16)f3.w;
  }

  #pragma unroll
  for (int w = 0; w < 2; ++w) {
    const _Float16* wt = Wt[w];
    float* Y = w ? Y2 : Y1;
    for (int j = 0; j < 8; ++j) {            // N-tiles
      v8f acc = {};
      #pragma unroll
      for (int kc = 0; kc < 4; ++kc) {
        // B fragment (col = m16): two contiguous 16-byte runs in transposed LDS
        const _Float16* wc =
            wt + (j * 16 + m16) * WT_STRIDE + kc * 32 + 8 * hi;
        const v8h b0 = *reinterpret_cast<const v8h*>(wc);        // K +0..7
        const v8h b1 = *reinterpret_cast<const v8h*>(wc + 16);   // K +16..23
        const v16h bfrag = __builtin_shufflevector(
            b0, b1, 0, 1, 2, 3, 4, 5, 6, 7, 8, 9, 10, 11, 12, 13, 14, 15);
        acc = __builtin_amdgcn_wmma_f32_16x16x32_f16(
            /*neg_a=*/false, afrag[kc], /*neg_b=*/false, bfrag,
            /*c_mod=*/(short)0, acc, /*reuse_a=*/false, /*reuse_b=*/false);
      }
      // D layout: VGPR r -> row = r + 8*hi, col = lane&15
      #pragma unroll
      for (int r = 0; r < 8; ++r) {
        Y[(size_t)(tileM * 16 + r + 8 * hi) * 128 + j * 16 + m16] = acc[r];
      }
    }
  }
}

// ------------------------------------------- per-(node,head) attention dots
__global__ __launch_bounds__(256)
void alpha_dots(const float* __restrict__ h, const float* __restrict__ att_l,
                const float* __restrict__ att_r, float* __restrict__ al,
                float* __restrict__ ar, long long n8) {
  long long gid = (long long)blockIdx.x * blockDim.x + threadIdx.x;
  if (gid >= n8) return;
  const long long node = gid >> 3;
  const int hd = (int)(gid & 7);
  const float* hp = h + node * 128 + hd * 16;
  const float* wl = att_l + hd * 16;
  const float* wr = att_r + hd * 16;
  float sl = 0.f, sr = 0.f;
  #pragma unroll
  for (int c = 0; c < 16; ++c) {
    const float v = hp[c];
    sl = fmaf(v, wl[c], sl);
    sr = fmaf(v, wr[c], sr);
  }
  al[gid] = sl;
  ar[gid] = sr;
}

// ------------------------------------- edge pass 1: logit + segment atomicMax
__global__ __launch_bounds__(256)
void edge_score_max(const int* __restrict__ esrc, const int* __restrict__ edst,
                    const float* __restrict__ al, const float* __restrict__ ar,
                    float* __restrict__ p, float* __restrict__ segmax,
                    long long e8) {
  long long gid = (long long)blockIdx.x * blockDim.x + threadIdx.x;
  if (gid >= e8) return;
  const long long e = gid >> 3;
  const int hd = (int)(gid & 7);
  const int s = esrc[e];
  const int d = edst[e];
  float a = al[(size_t)s * 8 + hd] + ar[(size_t)d * 8 + hd];
  a = a > 0.f ? a : LEAKY * a;  // LeakyReLU
  p[gid] = a;
  atomicMaxFloat(segmax + (size_t)d * 8 + hd, a);
}

// ------------------------------------- edge pass 2: exp + segment atomicAdd
__global__ __launch_bounds__(256)
void edge_exp_sum(const int* __restrict__ edst, float* __restrict__ p,
                  const float* __restrict__ segmax, float* __restrict__ segsum,
                  long long e8) {
  long long gid = (long long)blockIdx.x * blockDim.x + threadIdx.x;
  if (gid >= e8) return;
  const long long e = gid >> 3;
  const int hd = (int)(gid & 7);
  const int d = edst[e];
  const float v = __expf(p[gid] - segmax[(size_t)d * 8 + hd]);
  p[gid] = v;
  atomicAdd(segsum + (size_t)d * 8 + hd, v);
}

// ---------------- edge pass 3: normalize + float4 gather + atomic scatter-add
// One wave per edge; lane owns 4 contiguous channels (one global_load_b128 +
// 4 global_atomic_add_f32). This is the HBM-bound hot loop (~1.6 GB traffic).
__global__ __launch_bounds__(256)
void edge_message(const int* __restrict__ esrc, const int* __restrict__ edst,
                  const float* __restrict__ h, const float* __restrict__ p,
                  const float* __restrict__ segsum, float* __restrict__ agg,
                  int E) {
  const int e = blockIdx.x * 8 + (threadIdx.x >> 5);
  if (e >= E) return;
  const int lane = threadIdx.x & 31;
  const int s = esrc[e];
  const int d = edst[e];
  const int head = lane >> 2;  // (4*lane)/16
  const float coef =
      p[(size_t)e * 8 + head] / (segsum[(size_t)d * 8 + head] + EPS_DEN);
  const float4 hv =
      reinterpret_cast<const float4*>(h + (size_t)s * 128)[lane];
  float* ap = agg + (size_t)d * 128 + lane * 4;
  atomicAdd(ap + 0, hv.x * coef);
  atomicAdd(ap + 1, hv.y * coef);
  atomicAdd(ap + 2, hv.z * coef);
  atomicAdd(ap + 3, hv.w * coef);
}

// ------------------------------------------------ finalize: ELU + residual
__global__ __launch_bounds__(256)
void finalize(float* __restrict__ out, const float* __restrict__ res,
              long long n) {
  long long i = (long long)blockIdx.x * blockDim.x + threadIdx.x;
  if (i >= n) return;
  float v = out[i];
  v = v > 0.f ? v : (__expf(v) - 1.f);  // ELU
  out[i] = v + res[i];
}

// ---------------------------------------------------------------- launcher
extern "C" void kernel_launch(void* const* d_in, const int* in_sizes, int n_in,
                              void* d_out, int out_size, void* d_ws,
                              size_t ws_size, hipStream_t stream) {
  (void)n_in; (void)out_size; (void)ws_size;
  const float* x     = (const float*)d_in[0];
  const int*   edges = (const int*)d_in[1];
  const float* W_lin = (const float*)d_in[2];
  const float* att_l = (const float*)d_in[3];
  const float* att_r = (const float*)d_in[4];
  const float* W_res = (const float*)d_in[5];

  const int N = in_sizes[0] / 128;
  const int E = in_sizes[1] / 2;
  const int* esrc = edges;
  const int* edst = edges + E;

  // workspace carving (floats): h | res | al | ar | segmax | segsum | p
  float* ws = (float*)d_ws;
  size_t o = 0;
  float* h    = ws + o; o += (size_t)N * 128;
  float* res  = ws + o; o += (size_t)N * 128;
  float* al   = ws + o; o += (size_t)N * 8;
  float* ar   = ws + o; o += (size_t)N * 8;
  float* smax = ws + o; o += (size_t)N * 8;
  float* ssum = ws + o; o += (size_t)N * 8;
  float* p    = ws + o; o += (size_t)E * 8;
  float* out  = (float*)d_out;  // doubles as the agg accumulator

  const long long n8   = (long long)N * 8;
  const long long nout = (long long)N * 128;
  const long long e8   = (long long)E * 8;

  // init accumulators
  fill_f32<<<(unsigned)((n8 + 255) / 256), 256, 0, stream>>>(
      smax, -__builtin_huge_valf(), n8);
  fill_f32<<<(unsigned)((n8 + 255) / 256), 256, 0, stream>>>(ssum, 0.0f, n8);
  fill_f32<<<(unsigned)((nout + 255) / 256), 256, 0, stream>>>(out, 0.0f, nout);

  // h = x @ W_lin ; res = x @ W_res  (fused, x read once, WMMA)
  const int mtiles = (N + 15) / 16;
  const unsigned gemm_blocks = (unsigned)((mtiles + 7) / 8);
  wmma_dual_gemm_128<<<gemm_blocks, 256, 0, stream>>>(x, W_lin, W_res, h, res, N);

  // attention dot products
  alpha_dots<<<(unsigned)((n8 + 255) / 256), 256, 0, stream>>>(
      h, att_l, att_r, al, ar, n8);

  // segment softmax over edges (grouped by dst)
  edge_score_max<<<(unsigned)((e8 + 255) / 256), 256, 0, stream>>>(
      esrc, edst, al, ar, p, smax, e8);
  edge_exp_sum<<<(unsigned)((e8 + 255) / 256), 256, 0, stream>>>(
      edst, p, smax, ssum, e8);

  // weighted message scatter-add into d_out
  edge_message<<<(unsigned)((E + 7) / 8), 256, 0, stream>>>(
      esrc, edst, h, p, ssum, out, E);

  // out = elu(agg) + residual
  finalize<<<(unsigned)((nout + 255) / 256), 256, 0, stream>>>(out, res, nout);
}
